// MultiHeadAttention_16690242913120
// MI455X (gfx1250) — matrix operand order repaired and checked by reading, not compiled, not run on hardware
//
#include <hip/hip_runtime.h>
#include <hip/hip_bf16.h>

// MHA: out = softmax(mask(QWq (KWk)^T / sqrt(dk))) (VWv) Wo
// All matmuls via v_wmma_f32_16x16x32_bf16 (bf16 in, fp32 accumulate).
// Flash-attention streaming; K tiles staged into LDS via the Tensor Data Mover.

typedef __bf16 bf16_t;
typedef __attribute__((ext_vector_type(16))) __bf16 v16bf;
typedef __attribute__((ext_vector_type(8)))  __bf16 v8bf;
typedef __attribute__((ext_vector_type(8)))  float  v8f;
typedef __attribute__((ext_vector_type(4))) unsigned int u32x4;
typedef __attribute__((ext_vector_type(8))) int i32x8;
typedef __attribute__((ext_vector_type(4))) int i32x4;

#define B_  4
#define S_  2048
#define D_  1024
#define H_  16
#define DK_ 64

__device__ __forceinline__ v8f zero8() {
  v8f z;
#pragma unroll
  for (int i = 0; i < 8; i++) z[i] = 0.0f;
  return z;
}

// A-matrix 16x32 bf16 fragment (ISA 7.12.2): lane row = lane&15,
// lanes 0-15 hold K = {0..7, 16..23}, lanes 16-31 hold K = {8..15, 24..31}.
__device__ __forceinline__ v16bf load_a_frag(const bf16_t* base, int row, int ld,
                                             int kbase, int half) {
  const bf16_t* p = base + (size_t)row * ld + kbase + half * 8;
  v8bf lo = *(const v8bf*)(p);
  v8bf hi = *(const v8bf*)(p + 16);
  v16bf r;
#pragma unroll
  for (int i = 0; i < 8; i++) { r[i] = lo[i]; r[i + 8] = hi[i]; }
  return r;
}

// B-matrix 32x16 bf16 fragment: lane col = lane&15, lanes 0-15 hold K = {0..7, 16..23},
// lanes 16-31 hold K = {8..15, 24..31} (matches the A fragment).
// `base` row `col` holds that B-column's K-run contiguously.
__device__ __forceinline__ v16bf load_b_frag(const bf16_t* base, int col, int ld,
                                             int kbase, int half) {
  const bf16_t* p = base + (size_t)col * ld + kbase + half * 8;
  v8bf lo = *(const v8bf*)(p);
  v8bf hi = *(const v8bf*)(p + 16);
  v16bf r;
#pragma unroll
  for (int i = 0; i < 8; i++) { r[i] = lo[i]; r[i + 8] = hi[i]; }
  return r;
}

// ---------------------------------------------------------------------------
// fp32 -> bf16 bulk convert (n must be a multiple of 8; all our tensors are)
// ---------------------------------------------------------------------------
__global__ void cvt_f32_bf16(const float* __restrict__ x, bf16_t* __restrict__ y,
                             size_t n) {
  size_t i = ((size_t)blockIdx.x * blockDim.x + threadIdx.x) * 8;
  if (i + 8 <= n) {
    float4 a = *(const float4*)(x + i);
    float4 b = *(const float4*)(x + i + 4);
    v8bf o;
    o[0] = (bf16_t)a.x; o[1] = (bf16_t)a.y; o[2] = (bf16_t)a.z; o[3] = (bf16_t)a.w;
    o[4] = (bf16_t)b.x; o[5] = (bf16_t)b.y; o[6] = (bf16_t)b.z; o[7] = (bf16_t)b.w;
    *(v8bf*)(y + i) = o;
  }
}

// ---------------------------------------------------------------------------
// Tiled bf16 WMMA GEMM:  C[M,N] = A[M,K] * W[K,N]
// Block 128x128, K-step 32, 128 threads = 4 waves, each wave a 64x64 tile
// (16 WMMA per k-step, 1:1 ds_load_b128 : wmma ratio).
// MODE 0: store bf16, head-split layout [B,H,S,64] (for Q/K/V projections)
// MODE 1: store fp32 row-major (final output projection)
// ---------------------------------------------------------------------------
template <int MODE>
__global__ __launch_bounds__(128) void gemm_bf16_wmma(
    const bf16_t* __restrict__ A, const bf16_t* __restrict__ W,
    bf16_t* __restrict__ outH, float* __restrict__ outF,
    int M, int Kd, int N) {
  __shared__ bf16_t As[128][48];   // 128x32 tile, row padded to 48 (96B stride)
  __shared__ bf16_t Bs[128][48];   // W tile transposed: [n][k]

  const int tid  = threadIdx.x;
  const int wid  = tid >> 5;
  const int lane = tid & 31;
  const int r    = lane & 15;
  const int half = lane >> 4;
  const int bm   = blockIdx.x;
  const int bn   = blockIdx.y;
  const int wm   = wid >> 1;       // 0..1 -> 64-row slice
  const int wn   = wid & 1;        // 0..1 -> 64-col slice

  v8f acc[4][4];
#pragma unroll
  for (int mt = 0; mt < 4; mt++)
#pragma unroll
    for (int nt = 0; nt < 4; nt++) acc[mt][nt] = zero8();

  for (int kb = 0; kb < Kd; kb += 32) {
    __syncthreads();
    // A tile 128x32: 512 chunks of 8 bf16, 4 per thread
#pragma unroll
    for (int i = 0; i < 4; i++) {
      int c = tid + i * 128;
      int ar = c >> 2, ak = (c & 3) * 8;
      v8bf t = *(const v8bf*)(A + (size_t)(bm * 128 + ar) * Kd + kb + ak);
      *(v8bf*)(&As[ar][ak]) = t;
    }
    // W tile 32x128, stored transposed into Bs[n][k]
#pragma unroll
    for (int i = 0; i < 4; i++) {
      int c = tid + i * 128;
      int kr = c >> 4, nn = (c & 15) * 8;
      v8bf t = *(const v8bf*)(W + (size_t)(kb + kr) * N + bn * 128 + nn);
#pragma unroll
      for (int e = 0; e < 8; e++) Bs[nn + e][kr] = t[e];
    }
    if (kb + 32 < Kd)  // speculative prefetch of the next A tile slice
      __builtin_prefetch(A + (size_t)(bm * 128 + tid) * Kd + kb + 32, 0, 1);
    __syncthreads();

    v16bf af[4];
#pragma unroll
    for (int mt = 0; mt < 4; mt++)
      af[mt] = load_a_frag(&As[0][0], wm * 64 + mt * 16 + r, 48, 0, half);
#pragma unroll
    for (int nt = 0; nt < 4; nt++) {
      v16bf bfg = load_b_frag(&Bs[0][0], wn * 64 + nt * 16 + r, 48, 0, half);
#pragma unroll
      for (int mt = 0; mt < 4; mt++)
        acc[mt][nt] = __builtin_amdgcn_wmma_f32_16x16x32_bf16(
            false, af[mt], false, bfg, (short)0, acc[mt][nt], false, false);
    }
  }

  // C/D layout: VGPR j holds row j (lanes 0-15) / row j+8 (lanes 16-31), col = lane&15
#pragma unroll
  for (int mt = 0; mt < 4; mt++)
#pragma unroll
    for (int nt = 0; nt < 4; nt++)
#pragma unroll
      for (int j = 0; j < 8; j++) {
        int row = bm * 128 + wm * 64 + mt * 16 + j + 8 * half;
        int col = bn * 128 + wn * 64 + nt * 16 + r;
        float v = acc[mt][nt][j];
        if (MODE == 0) {
          int b = row >> 11, s = row & (S_ - 1);
          int h = col >> 6,  d = col & 63;
          outH[(((size_t)(b * H_ + h) * S_ + s) << 6) + d] = (bf16_t)v;
        } else {
          outF[(size_t)row * N + col] = v;
        }
      }
}

// ---------------------------------------------------------------------------
// Flash attention: block = one (b,h) x 64 queries, 4 waves x 16 queries each.
// Streams S/64 key tiles; online softmax in fp32; both matmuls via WMMA bf16.
// K tile is DMA'd into LDS by the Tensor Data Mover (with LDS row padding);
// V tile is loaded+transposed manually (TDM cannot transpose).
// ---------------------------------------------------------------------------
__global__ __launch_bounds__(128) void flash_attn_wmma(
    const bf16_t* __restrict__ qh, const bf16_t* __restrict__ kh,
    const bf16_t* __restrict__ vh, const unsigned char* __restrict__ mask,
    bf16_t* __restrict__ ctx) {
  __shared__ bf16_t Ks[64][72];       // key tile, row-major [key][dk], pad 8
  __shared__ bf16_t Vt[64][72];       // V tile transposed [dv][key]
  __shared__ bf16_t Ps[4][16][72];    // per-wave P staging [q][key]

  const int tid  = threadIdx.x;
  const int wid  = tid >> 5;
  const int lane = tid & 31;
  const int r    = lane & 15;
  const int half = lane >> 4;
  const int bh   = blockIdx.y;
  const int b    = bh >> 4;
  const int h    = bh & 15;
  const int q0   = blockIdx.x * 64;
  const size_t headoff = (size_t)(b * H_ + h) * S_ * DK_;

  // persistent Q fragments for this wave's 16 queries (dk = 2 k-steps of 32)
  const bf16_t* qbase = qh + headoff + (size_t)(q0 + wid * 16) * DK_;
  v16bf qf[2];
  qf[0] = load_a_frag(qbase, r, DK_, 0, half);
  qf[1] = load_a_frag(qbase, r, DK_, 32, half);

  float mrow[8], lrow[8];
#pragma unroll
  for (int j = 0; j < 8; j++) { mrow[j] = -3.0e38f; lrow[j] = 0.0f; }
  v8f O[4];
#pragma unroll
  for (int nt = 0; nt < 4; nt++) O[nt] = zero8();

  const unsigned char* mbase = mask + (size_t)b * S_ * S_;
  // LDS aperture: flat address low 32 bits map to the LDS offset
  const unsigned ks_lds = (unsigned)(size_t)(&Ks[0][0]);

  for (int kt = 0; kt < S_ / 64; kt++) {
    __syncthreads();
    const bf16_t* kbp = kh + headoff + (size_t)kt * 64 * DK_;
    const bf16_t* vbp = vh + headoff + (size_t)kt * 64 * DK_;

    // --- K tile via Tensor Data Mover (one wave issues the DMA) ---
    // D# group0: count=1 | lds_addr | 57b global addr | type=2 ("image")
    // D# group1: data_size=2B, pad_enable, pad_interval=32 DW (one 128B row),
    //            pad_amount=4 DW (16B row pad); tensor/tile 64x64, stride 64.
    if (wid == 0) {
      unsigned long long ga = (unsigned long long)(size_t)kbp;
      u32x4 g0 = { 1u, ks_lds,
                   (unsigned)(ga & 0xffffffffu),
                   (unsigned)((ga >> 32) & 0x01ffffffu) | (2u << 30) };
      i32x8 g1 = { (int)0x07110000,            // data_size=1(2B) pad_en int=4 amt=3
                   (int)((64u & 0xffffu) << 16),  // tensor_dim0 lo16
                   (int)((64u & 0xffffu) << 16),  // tensor_dim0 hi=0, tensor_dim1 lo16
                   (int)(64u << 16),              // tensor_dim1 hi=0, tile_dim0=64
                   (int)64,                       // tile_dim1=64, tile_dim2=0
                   (int)64,                       // tensor_dim0_stride lo32
                   0, 0 };
      i32x4 gz  = { 0, 0, 0, 0 };
      i32x8 gz8 = { 0, 0, 0, 0, 0, 0, 0, 0 };
      __builtin_amdgcn_tensor_load_to_lds(g0, g1, gz, gz, gz8, 0);
    }

    // --- V tile loaded + transposed into Vt: 512 8-elt chunks, 4 per thread ---
#pragma unroll
    for (int i = 0; i < 4; i++) {
      int c = tid + i * 128;
      int krow = c >> 3, kk = (c & 7) * 8;
      v8bf tv = *(const v8bf*)(vbp + (size_t)krow * DK_ + kk);
#pragma unroll
      for (int e = 0; e < 8; e++) Vt[kk + e][krow] = tv[e];
    }
    if (wid == 0) __builtin_amdgcn_s_wait_tensorcnt(0);
    __syncthreads();

    // scores S = Q K^T : 4 key n-tiles x 2 dk k-steps
    v8f st[4];
#pragma unroll
    for (int t4 = 0; t4 < 4; t4++) st[t4] = zero8();
#pragma unroll
    for (int t4 = 0; t4 < 4; t4++)
#pragma unroll
      for (int kk2 = 0; kk2 < 2; kk2++) {
        v16bf bk = load_b_frag(&Ks[0][0], t4 * 16 + r, 72, kk2 * 32, half);
        st[t4] = __builtin_amdgcn_wmma_f32_16x16x32_bf16(
            false, qf[kk2], false, bk, (short)0, st[t4], false, false);
      }

    // scale + mask (mask==true -> -1e9, matching reference)
#pragma unroll
    for (int t4 = 0; t4 < 4; t4++)
#pragma unroll
      for (int j = 0; j < 8; j++) {
        float s = st[t4][j] * 0.125f;  // 1/sqrt(64)
        int qr = q0 + wid * 16 + j + 8 * half;
        int kc = kt * 64 + t4 * 16 + r;
        if (mbase[(size_t)qr * S_ + kc]) s = -1.0e9f;
        st[t4][j] = s;
      }

    // online softmax: row = j + 8*half, spread over 16 lanes (one col each)
    float mnew[8], alpha[8], rs[8];
#pragma unroll
    for (int j = 0; j < 8; j++) {
      float mx = fmaxf(fmaxf(st[0][j], st[1][j]), fmaxf(st[2][j], st[3][j]));
#pragma unroll
      for (int d = 1; d < 16; d <<= 1) mx = fmaxf(mx, __shfl_xor(mx, d, 32));
      mnew[j]  = fmaxf(mrow[j], mx);
      alpha[j] = __expf(mrow[j] - mnew[j]);
      rs[j]    = 0.0f;
    }
#pragma unroll
    for (int t4 = 0; t4 < 4; t4++)
#pragma unroll
      for (int j = 0; j < 8; j++) {
        float p = __expf(st[t4][j] - mnew[j]);
        st[t4][j] = p;
        rs[j] += p;
      }
#pragma unroll
    for (int j = 0; j < 8; j++) {
#pragma unroll
      for (int d = 1; d < 16; d <<= 1) rs[j] += __shfl_xor(rs[j], d, 32);
      lrow[j] = lrow[j] * alpha[j] + rs[j];
      mrow[j] = mnew[j];
    }
#pragma unroll
    for (int nt = 0; nt < 4; nt++)
#pragma unroll
      for (int j = 0; j < 8; j++) O[nt][j] *= alpha[j];

    // P: C-layout regs -> LDS row-major bf16 (per-wave region, LDS is in-order)
    bf16_t* Pw = &Ps[wid][0][0];
#pragma unroll
    for (int t4 = 0; t4 < 4; t4++)
#pragma unroll
      for (int j = 0; j < 8; j++)
        Pw[(j + 8 * half) * 72 + t4 * 16 + r] = (bf16_t)st[t4][j];

    // O += P V : 4 dv n-tiles x 2 key k-steps
#pragma unroll
    for (int nt = 0; nt < 4; nt++)
#pragma unroll
      for (int kk2 = 0; kk2 < 2; kk2++) {
        v16bf pa = load_a_frag(Pw, r, 72, kk2 * 32, half);
        v16bf bv = load_b_frag(&Vt[0][0], nt * 16 + r, 72, kk2 * 32, half);
        O[nt] = __builtin_amdgcn_wmma_f32_16x16x32_bf16(
            false, pa, false, bv, (short)0, O[nt], false, false);
      }
  }

  // normalize and store ctx in [B, S, H*64] bf16 for the output projection
#pragma unroll
  for (int nt = 0; nt < 4; nt++)
#pragma unroll
    for (int j = 0; j < 8; j++) {
      int qr = q0 + wid * 16 + j + 8 * half;
      int dc = nt * 16 + r;
      float o = O[nt][j] / lrow[j];
      ctx[(size_t)(b * S_ + qr) * (H_ * DK_) + h * 64 + dc] = (bf16_t)o;
    }
}

// ---------------------------------------------------------------------------
extern "C" void kernel_launch(void* const* d_in, const int* in_sizes, int n_in,
                              void* d_out, int out_size, void* d_ws, size_t ws_size,
                              hipStream_t stream) {
  const float* Q  = (const float*)d_in[0];
  const float* K  = (const float*)d_in[1];
  const float* V  = (const float*)d_in[2];
  const unsigned char* mask = (const unsigned char*)d_in[3];  // jax bool_ = 1 byte
  const float* WQ = (const float*)d_in[4];
  const float* WK = (const float*)d_in[5];
  const float* WV = (const float*)d_in[6];
  const float* WO = (const float*)d_in[7];
  float* out = (float*)d_out;

  char* ws = (char*)d_ws;
  size_t off = 0;
  auto alloc = [&](size_t bytes) -> bf16_t* {
    bf16_t* p = (bf16_t*)(ws + off);
    off += (bytes + 255) & ~(size_t)255;
    return p;
  };
  const size_t MS = (size_t)B_ * S_;  // 8192 rows
  bf16_t* Qb  = alloc(MS * D_ * 2);
  bf16_t* Kb  = alloc(MS * D_ * 2);
  bf16_t* Vb  = alloc(MS * D_ * 2);
  bf16_t* WQb = alloc((size_t)D_ * D_ * 2);
  bf16_t* WKb = alloc((size_t)D_ * D_ * 2);
  bf16_t* WVb = alloc((size_t)D_ * D_ * 2);
  bf16_t* WOb = alloc((size_t)D_ * D_ * 2);
  bf16_t* qhp = alloc(MS * D_ * 2);   // [B,H,S,64]
  bf16_t* khp = alloc(MS * D_ * 2);
  bf16_t* vhp = alloc(MS * D_ * 2);
  bf16_t* ctx = alloc(MS * D_ * 2);   // [B,S,H*64]

  auto launch_cvt = [&](const float* src, bf16_t* dst, size_t n) {
    cvt_f32_bf16<<<dim3((unsigned)(n / 8 / 256)), dim3(256), 0, stream>>>(src, dst, n);
  };
  launch_cvt(Q,  Qb,  MS * D_);
  launch_cvt(K,  Kb,  MS * D_);
  launch_cvt(V,  Vb,  MS * D_);
  launch_cvt(WQ, WQb, (size_t)D_ * D_);
  launch_cvt(WK, WKb, (size_t)D_ * D_);
  launch_cvt(WV, WVb, (size_t)D_ * D_);
  launch_cvt(WO, WOb, (size_t)D_ * D_);

  dim3 gemm_grid(64, 8), gemm_blk(128);
  gemm_bf16_wmma<0><<<gemm_grid, gemm_blk, 0, stream>>>(Qb, WQb, qhp, nullptr, 8192, 1024, 1024);
  gemm_bf16_wmma<0><<<gemm_grid, gemm_blk, 0, stream>>>(Kb, WKb, khp, nullptr, 8192, 1024, 1024);
  gemm_bf16_wmma<0><<<gemm_grid, gemm_blk, 0, stream>>>(Vb, WVb, vhp, nullptr, 8192, 1024, 1024);

  flash_attn_wmma<<<dim3(S_ / 64, B_ * H_), dim3(128), 0, stream>>>(qhp, khp, vhp, mask, ctx);

  gemm_bf16_wmma<1><<<gemm_grid, gemm_blk, 0, stream>>>(ctx, WOb, nullptr, out, 8192, 1024, 1024);
}
